// GAT_OneRoud_13322988552243
// MI455X (gfx1250) — matrix operand order repaired and checked by reading, not compiled, not run on hardware
//
#include <hip/hip_runtime.h>
#include <hip/hip_bf16.h>
#include <stdint.h>

typedef __attribute__((ext_vector_type(16))) __bf16 bf16x16;
typedef __attribute__((ext_vector_type(8)))  __bf16 bf16x8;
typedef __attribute__((ext_vector_type(8)))  float  f32x8;

#define HCDIM 128
#define HEADS 4
#define CH 32
#define NEG_SLOPE 0.2f
#define EPSV 1e-16f
#define NEGINF_KEY 0x007FFFFFu  // monotone key of -inf

__device__ __forceinline__ unsigned short f2bf(float f) {
    unsigned u = __float_as_uint(f);
    unsigned r = 0x7fffu + ((u >> 16) & 1u);   // round-to-nearest-even
    return (unsigned short)((u + r) >> 16);
}
__device__ __forceinline__ unsigned fkey(float f) {   // order-preserving uint key
    unsigned u = __float_as_uint(f);
    return (u & 0x80000000u) ? ~u : (u | 0x80000000u);
}
__device__ __forceinline__ float keyf(unsigned k) {
    unsigned u = (k & 0x80000000u) ? (k & 0x7fffffffu) : ~k;
    return __uint_as_float(u);
}

// ---------------- elementwise converts ----------------
__global__ void k_f32_to_bf16(const float* __restrict__ in,
                              unsigned short* __restrict__ out, int n) {
    int i = blockIdx.x * blockDim.x + threadIdx.x;
    if (i < n) out[i] = f2bf(in[i]);
}

// W [128(K) x 128(col)] f32 row-major  ->  Wt [col][K] bf16 (column-major)
__global__ void k_w_transpose_bf16(const float* __restrict__ W,
                                   unsigned short* __restrict__ Wt) {
    int i = blockIdx.x * blockDim.x + threadIdx.x;   // 0..16383
    int k = i >> 7, c = i & 127;
    Wt[(size_t)c * HCDIM + k] = f2bf(W[(size_t)k * HCDIM + c]);
}

// H[n,c] = b[0][c] + b[1][c] + b[2][c]  (HeteroConv aggr='sum' bias fold)
__global__ void k_bias_init(float* __restrict__ H, const float* __restrict__ b3,
                            long total) {
    long i = (long)blockIdx.x * blockDim.x + threadIdx.x;
    if (i >= total) return;
    int c = (int)(i & (HCDIM - 1));
    H[i] = b3[c] + b3[HCDIM + c] + b3[2 * HCDIM + c];
}

__global__ void k_seg_init(unsigned* __restrict__ nmax, float* __restrict__ denom, int n4) {
    int i = blockIdx.x * blockDim.x + threadIdx.x;
    if (i < n4) { nmax[i] = NEGINF_KEY; denom[i] = 0.f; }
}

// ---------------- fused dual WMMA GEMM ----------------
// YL = Xbf @ Wl, YR = Xbf @ Wr  (Wt buffers are col-major bf16).
// grid.x = ceil(N/16) row tiles, block = 256 (8 waves), wave w -> cols 16w..16w+15.
// Shared A fragment feeds two independent accumulator chains (8 v_wmma / wave).
__global__ void k_gemm2_bf16_wmma(const unsigned short* __restrict__ Xu,
                                  const unsigned short* __restrict__ WtLu,
                                  const unsigned short* __restrict__ WtRu,
                                  float* __restrict__ YL, float* __restrict__ YR,
                                  int N) {
    const __bf16* X   = (const __bf16*)Xu;
    const __bf16* WtL = (const __bf16*)WtLu;
    const __bf16* WtR = (const __bf16*)WtRu;
    int wave = threadIdx.x >> 5;
    int lane = threadIdx.x & 31;
    int m    = lane & 15;
    bool hi  = lane >= 16;
    int row0 = blockIdx.x * 16;
    int col0 = wave * 16;

    int rA = row0 + m; if (rA >= N) rA = N - 1;            // clamp for tail safety
    const __bf16* arow  = X   + (size_t)rA * HCDIM;        // A row (M = lane&15)
    const __bf16* bcolL = WtL + (size_t)(col0 + m) * HCDIM; // B col (N = lane&15)
    const __bf16* bcolR = WtR + (size_t)(col0 + m) * HCDIM;

    f32x8 accL = {}, accR = {};
#pragma unroll
    for (int kk = 0; kk < HCDIM; kk += 32) {
        // A 16x32 bf16 fragment (contiguous K, matching B): lanes<16 hold K=kk..kk+15,
        // lanes>=16 hold K=kk+16..kk+31.
        int ka0 = kk + (hi ? 16 : 0);
        int ka1 = ka0 + 8;
        bf16x8 alo = *(const bf16x8*)(arow + ka0);
        bf16x8 ahi = *(const bf16x8*)(arow + ka1);
        bf16x16 a;
#pragma unroll
        for (int i = 0; i < 8; ++i) { a[i] = alo[i]; a[i + 8] = ahi[i]; }
        // B 32x16 bf16 fragment: lanes<16 hold K=kk..kk+15, lanes>=16 K=kk+16..kk+31
        int kb = kk + (hi ? 16 : 0);
        bf16x16 bl = *(const bf16x16*)(bcolL + kb);
        bf16x16 br = *(const bf16x16*)(bcolR + kb);
        accL = __builtin_amdgcn_wmma_f32_16x16x32_bf16(
                   false, a, false, bl, (short)0, accL, false, false);
        accR = __builtin_amdgcn_wmma_f32_16x16x32_bf16(
                   false, a, false, br, (short)0, accR, false, false);
    }
    // D 16x16 f32: VGPR j -> M = j (+8 for hi lanes), N = lane&15
    size_t base = (size_t)(row0 + (hi ? 8 : 0)) * HCDIM + col0 + m;
    if (row0 + 16 <= N) {                       // block-uniform fast path (no branches)
#pragma unroll
        for (int j = 0; j < 8; ++j) YL[base + (size_t)j * HCDIM] = accL[j];
#pragma unroll
        for (int j = 0; j < 8; ++j) YR[base + (size_t)j * HCDIM] = accR[j];
    } else {
#pragma unroll
        for (int j = 0; j < 8; ++j) {
            int rowj = row0 + j + (hi ? 8 : 0);
            if (rowj < N) {
                YL[(size_t)rowj * HCDIM + col0 + m] = accL[j];
                YR[(size_t)rowj * HCDIM + col0 + m] = accR[j];
            }
        }
    }
}

// ---------------- edge kernels (wave32 per edge) ----------------
// e[e,h] = att[h,:] . leaky_relu(xl[src] + xr[dst]) ; atomic segment-max on dst
__global__ void k_edge_logits(const float* __restrict__ XL, const float* __restrict__ XR,
                              const int* __restrict__ src, const int* __restrict__ dst,
                              const float* __restrict__ att, float* __restrict__ elog,
                              unsigned* __restrict__ nmax, int E) {
    int eid  = blockIdx.x * (blockDim.x >> 5) + (threadIdx.x >> 5);
    int lane = threadIdx.x & 31;
    if (eid >= E) return;
    int s = src[eid], d = dst[eid];
    const float* pl = XL + (size_t)s * HCDIM;
    const float* pr = XR + (size_t)d * HCDIM;
    float p[HEADS];
#pragma unroll
    for (int h = 0; h < HEADS; ++h) {
        int idx = h * CH + lane;                  // head h, channel = lane
        float v = pl[idx] + pr[idx];
        v = v > 0.f ? v : NEG_SLOPE * v;
        p[h] = v * att[idx];
    }
#pragma unroll
    for (int off = 16; off; off >>= 1) {
#pragma unroll
        for (int h = 0; h < HEADS; ++h) p[h] += __shfl_xor(p[h], off, 32);
    }
    if (lane == 0) {
#pragma unroll
        for (int h = 0; h < HEADS; ++h) {
            elog[(size_t)eid * HEADS + h] = p[h];
            atomicMax(&nmax[(size_t)d * HEADS + h], fkey(p[h]));
        }
    }
}

// ex = exp(e - max[dst]) in place; denom[dst,h] += ex
__global__ void k_edge_exp(float* __restrict__ elog, const int* __restrict__ dst,
                           const unsigned* __restrict__ nmax, float* __restrict__ denom,
                           int E) {
    int t = blockIdx.x * blockDim.x + threadIdx.x;
    if (t >= E * HEADS) return;
    int eid = t >> 2, h = t & 3;
    int d = dst[eid];
    float mx = keyf(nmax[(size_t)d * HEADS + h]);
    float ex = __expf(elog[t] - mx);
    elog[t] = ex;
    atomicAdd(&denom[(size_t)d * HEADS + h], ex);
}

// H[dst] += (ex/(denom+eps)) * xl[src]   (float global atomics, wave per edge)
__global__ void k_scatter(const float* __restrict__ XL, const int* __restrict__ src,
                          const int* __restrict__ dst, const float* __restrict__ elog,
                          const float* __restrict__ denom, float* __restrict__ H, int E) {
    int eid  = blockIdx.x * (blockDim.x >> 5) + (threadIdx.x >> 5);
    int lane = threadIdx.x & 31;
    if (eid >= E) return;
    int s = src[eid], d = dst[eid];
    float alpha[HEADS];
#pragma unroll
    for (int h = 0; h < HEADS; ++h)
        alpha[h] = elog[(size_t)eid * HEADS + h] / (denom[(size_t)d * HEADS + h] + EPSV);
    const float* pl = XL + (size_t)s * HCDIM;
    float*       po = H  + (size_t)d * HCDIM;
#pragma unroll
    for (int h = 0; h < HEADS; ++h) {
        int idx = h * CH + lane;
        atomicAdd(&po[idx], alpha[h] * pl[idx]);
    }
}

// ---------------- final classifier ----------------
// out[g,o] = fcb[o] + sum_k relu(H[g*512+k]) * fcW[k,o]
__global__ void k_fc(const float* __restrict__ H, const float* __restrict__ fcW,
                     const float* __restrict__ fcb, float* __restrict__ out, int G) {
    int t = blockIdx.x * blockDim.x + threadIdx.x;
    if (t >= G * 4) return;
    int g = t >> 2, o = t & 3;
    const float* hrow = H + (size_t)g * 512;
    float acc = fcb[o];
    for (int k = 0; k < 512; ++k) {
        float v = hrow[k];
        v = v > 0.f ? v : 0.f;
        acc = fmaf(v, fcW[k * 4 + o], acc);
    }
    out[t] = acc;
}

// ---------------- host orchestration ----------------
static void run_layer(const unsigned short* Xbf, const unsigned short* WtL,
                      const int* const eptr[3], const float* att, const float* b3,
                      float* XL, float* XR, float* Hacc, float* elog,
                      unsigned* nmax, float* denom, int N, int E, hipStream_t stream) {
    long totalH = (long)N * HCDIM;
    k_bias_init<<<(unsigned)((totalH + 255) / 256), 256, 0, stream>>>(Hacc, b3, totalH);
    int rowTiles = (N + 15) / 16;
    int eblocks  = (E + 7) / 8;
    const size_t WMAT = (size_t)HCDIM * HCDIM;
    for (int r = 0; r < 3; ++r) {
        k_gemm2_bf16_wmma<<<rowTiles, 256, 0, stream>>>(
            Xbf, WtL + (size_t)(r * 2 + 0) * WMAT, WtL + (size_t)(r * 2 + 1) * WMAT,
            XL, XR, N);
        k_seg_init<<<(N * 4 + 255) / 256, 256, 0, stream>>>(nmax, denom, N * 4);
        k_edge_logits<<<eblocks, 256, 0, stream>>>(
            XL, XR, eptr[r], eptr[r] + E, att + (size_t)r * HEADS * CH, elog, nmax, E);
        k_edge_exp<<<(E * 4 + 255) / 256, 256, 0, stream>>>(elog, eptr[r] + E, nmax, denom, E);
        k_scatter<<<eblocks, 256, 0, stream>>>(XL, eptr[r], eptr[r] + E, elog, denom, Hacc, E);
    }
}

extern "C" void kernel_launch(void* const* d_in, const int* in_sizes, int n_in,
                              void* d_out, int out_size, void* d_ws, size_t ws_size,
                              hipStream_t stream) {
    (void)n_in; (void)out_size; (void)ws_size;
    const float* x    = (const float*)d_in[0];
    const int*   ef   = (const int*)d_in[1];
    const int*   ea   = (const int*)d_in[2];
    const int*   ev   = (const int*)d_in[3];
    const float* W1l  = (const float*)d_in[4];
    const float* W1r  = (const float*)d_in[5];
    const float* att1 = (const float*)d_in[6];
    const float* b1   = (const float*)d_in[7];
    const float* W2l  = (const float*)d_in[8];
    const float* W2r  = (const float*)d_in[9];
    const float* att2 = (const float*)d_in[10];
    const float* b2   = (const float*)d_in[11];
    const float* fcW  = (const float*)d_in[12];
    const float* fcb  = (const float*)d_in[13];
    float* out = (float*)d_out;

    int N = in_sizes[0] / HCDIM;   // F == HC == 128
    int E = in_sizes[1] / 2;
    const size_t WMAT = (size_t)HCDIM * HCDIM;  // 16384 elements

    char* ws = (char*)d_ws;
    size_t off = 0;
    auto carve = [&](size_t bytes) -> void* {
        void* p = ws + off;
        off = (off + bytes + 255) & ~(size_t)255;
        return p;
    };
    unsigned short* Xbf = (unsigned short*)carve((size_t)N * HCDIM * 2);
    unsigned short* Wt  = (unsigned short*)carve(12 * WMAT * 2);
    float*    XL    = (float*)carve((size_t)N * HCDIM * 4);
    float*    XR    = (float*)carve((size_t)N * HCDIM * 4);
    float*    Hacc  = (float*)carve((size_t)N * HCDIM * 4);
    float*    elog  = (float*)carve((size_t)E * HEADS * 4);
    unsigned* nmax  = (unsigned*)carve((size_t)N * HEADS * 4);
    float*    denom = (float*)carve((size_t)N * HEADS * 4);

    int nx = N * HCDIM;
    k_f32_to_bf16<<<(nx + 255) / 256, 256, 0, stream>>>(x, Xbf, nx);
    for (int r = 0; r < 3; ++r) {  // 12 weight transposes: layer1 idx 0..5, layer2 idx 6..11
        k_w_transpose_bf16<<<64, 256, 0, stream>>>(W1l + r * WMAT, Wt + (size_t)(r * 2 + 0) * WMAT);
        k_w_transpose_bf16<<<64, 256, 0, stream>>>(W1r + r * WMAT, Wt + (size_t)(r * 2 + 1) * WMAT);
        k_w_transpose_bf16<<<64, 256, 0, stream>>>(W2l + r * WMAT, Wt + (size_t)(6 + r * 2 + 0) * WMAT);
        k_w_transpose_bf16<<<64, 256, 0, stream>>>(W2r + r * WMAT, Wt + (size_t)(6 + r * 2 + 1) * WMAT);
    }
    const int* eptr[3] = {ef, ea, ev};

    run_layer(Xbf, Wt, eptr, att1, b1, XL, XR, Hacc, elog, nmax, denom, N, E, stream);
    k_f32_to_bf16<<<(nx + 255) / 256, 256, 0, stream>>>(Hacc, Xbf, nx);  // H1 -> bf16
    run_layer(Xbf, Wt + 6 * WMAT, eptr, att2, b2, XL, XR, Hacc, elog, nmax, denom, N, E, stream);

    int G = N / 4;
    k_fc<<<(G * 4 + 255) / 256, 256, 0, stream>>>(Hacc, fcW, fcb, out, G);
}
